// Network_60593398612490
// MI455X (gfx1250) — compile-verified, matrix operand order checked by reading, not run on hardware
//
#include <hip/hip_runtime.h>

// ---------------- problem constants ----------------
#define NE_  8192
#define NI_  2048
#define NN_  10240       // NE + NI
#define BB_  16
#define TT_  100

#define C_DT        0.01f
#define C_USE       0.03f
#define C_DT_FAC    0.01f          // DT / TAU_FAC
#define C_DT_REC    0.04f          // DT / TAU_REC
#define C_EXP_SYN   0.60653065971f // exp(-0.5)
#define C_DT_SYN    0.5f
#define C_EXP_NMDA  0.95122942450f // exp(-0.05)
#define C_RN_DTN    0.02f          // R_NMDA * DT_NMDA = 0.4 * 0.05
#define C_EXP_TAU   0.60653065971f
#define C_DT_TAU    0.5f
#define C_THRESH    1.0f

// ---------------- types ----------------
typedef __attribute__((ext_vector_type(16))) __bf16       v16bf;
typedef __attribute__((ext_vector_type(8)))  float        v8f;
typedef __attribute__((ext_vector_type(4)))  unsigned int u32x4;

__device__ __forceinline__ unsigned short f32_to_bf16(float f) {
    unsigned int u = __float_as_uint(f);
    u = (u + 0x7FFFu + ((u >> 16) & 1u)) >> 16;   // round-to-nearest-even
    return (unsigned short)u;
}

// Load one 16x32 bf16 WMMA operand fragment for this lane (regular temporal:
// used for A operands, which live in L2 and are re-read by every wave).
__device__ __forceinline__ v16bf load_frag(const unsigned short* p) {
    union { v16bf v; u32x4 q[2]; } f;
    f.q[0] = *(const u32x4*)(p);
    f.q[1] = *(const u32x4*)(p + 16);
    return f.v;
}

// Non-temporal fragment load: B operands (weights) stream 344 MB/step,
// bigger than the 192 MB L2 — don't let them evict the resident A operands.
__device__ __forceinline__ v16bf load_frag_nt(const unsigned short* p) {
    union { v16bf v; u32x4 q[2]; } f;
    f.q[0] = __builtin_nontemporal_load((const u32x4*)(p));
    f.q[1] = __builtin_nontemporal_load((const u32x4*)(p + 16));
    return f.v;
}

// ---------------- once-per-launch: fp32 row-major -> bf16 column-major ----------------
__global__ void transpose_to_bf16(const float* __restrict__ in,
                                  unsigned short* __restrict__ out, int dim) {
    __shared__ float tile[32][33];
    const int tx = threadIdx.x, ty = threadIdx.y;      // block (32,8)
    const int x0 = blockIdx.x * 32, y0 = blockIdx.y * 32;
#pragma unroll
    for (int r = 0; r < 32; r += 8)
        tile[ty + r][tx] = in[(size_t)(y0 + ty + r) * dim + (x0 + tx)];
    __syncthreads();
#pragma unroll
    for (int r = 0; r < 32; r += 8)
        out[(size_t)(x0 + ty + r) * dim + (y0 + tx)] = f32_to_bf16(tile[tx][ty + r]);
}

// ---------------- once-per-launch: initial state ----------------
__global__ void init_state(const float* __restrict__ ff, const float* __restrict__ rec,
                           float* __restrict__ rec0, float* __restrict__ rec1,
                           float* __restrict__ ratesf, unsigned short* __restrict__ ratesb,
                           float* __restrict__ u, float* __restrict__ x) {
    const int idx = blockIdx.x * blockDim.x + threadIdx.x;   // over B*N
    if (idx >= BB_ * NN_) return;
    const int b = idx / NN_, j = idx - b * NN_;
    const float r0 = rec[idx];
    const float r1 = rec[BB_ * NN_ + idx];
    rec0[idx] = r0;
    rec1[idx] = r1;
    float v = ff[(size_t)b * TT_ * NN_ + j] + r0 - C_THRESH;   // ff_input[:,0]
    v = v > 0.f ? v : 0.f;
    ratesf[idx] = v;
    ratesb[idx] = f32_to_bf16(v);
    if (j < NE_) { const int ui = b * NE_ + j; u[ui] = C_USE; x[ui] = 1.0f; }
}

// ---------------- per-step: STP variables + aux (bf16 A-operand) ----------------
__global__ void stp_update(const float* __restrict__ ratesf,
                           float* __restrict__ u, float* __restrict__ x,
                           unsigned short* __restrict__ aux) {
    const int idx = blockIdx.x * blockDim.x + threadIdx.x;   // over B*NE
    const int b = idx >> 13, e = idx & (NE_ - 1);
    const float re = ratesf[(size_t)b * NN_ + e];
    float uu = u[idx], xx = x[idx];
    uu = uu + C_DT_FAC * (C_USE - uu) + (C_USE * C_DT) * (1.0f - uu) * re;
    xx = xx + C_DT_REC * (1.0f - xx) - C_DT * uu * xx * re;
    u[idx] = uu;
    x[idx] = xx;
    aux[idx] = f32_to_bf16(uu * xx * re);
}

// ---------------- per-step: fused WMMA GEMMs + state update ----------------
// 4 waves cooperate on one 16-neuron output tile (M=16 = batch = one WMMA tile).
// Each wave takes 1/4 of each K-range (main-E: k<NE, main-I: NE<=k<N, STP: e<NE),
// partials combine through LDS, wave 0 runs the fused state-update epilogue.
// hidden2 (re @ Wab_E) == the k<NE partial of hidden (rates @ Wab_T).
__global__ __launch_bounds__(128) void step_gemm(
    const unsigned short* __restrict__ ratesb,   // (B,N) bf16, A of main GEMM
    const unsigned short* __restrict__ aux,      // (B,NE) bf16, A of STP GEMM
    const unsigned short* __restrict__ WtMain,   // (N,N) bf16 col-major: [j*N + k]
    const unsigned short* __restrict__ WtStp,    // (NE,NE) bf16 col-major: [j*NE + e]
    float* __restrict__ rec0, float* __restrict__ rec1,
    const float* __restrict__ ratesf_in, const float* __restrict__ ff, int t,
    float* __restrict__ ratesf_out, unsigned short* __restrict__ ratesb_out,
    float* __restrict__ out)
{
    __shared__ __align__(32) float red[3][3][256];   // [wave-1][accum][lane*8+i]

    const int tid  = threadIdx.x;
    const int wave = tid >> 5;
    const int lane = tid & 31;
    const int j0   = blockIdx.x << 4;           // output-neuron tile base
    const int mn   = lane & 15;                 // A row (m) == B column (n)
    const int kofs = (lane >> 4) << 3;          // per-lane K base offset

    const unsigned short* Arow = ratesb + (size_t)mn * NN_ + kofs;
    const unsigned short* Brow = WtMain + (size_t)(j0 + mn) * NN_ + kofs;

    // Per-wave K ranges (elements): main-E 2048, main-I 512, STP 2048.
    const int eBeg = wave * (NE_ / 4), eEnd = eBeg + NE_ / 4;
    const int iBeg = NE_ + wave * (NI_ / 4), iEnd = iBeg + NI_ / 4;

    v8f accE = {};
#pragma unroll 8
    for (int k = eBeg; k < eEnd; k += 32)
        accE = __builtin_amdgcn_wmma_f32_16x16x32_bf16(
            false, load_frag(Arow + k), false, load_frag_nt(Brow + k),
            (short)0, accE, false, false);

    v8f accI = {};
#pragma unroll 4
    for (int k = iBeg; k < iEnd; k += 32)
        accI = __builtin_amdgcn_wmma_f32_16x16x32_bf16(
            false, load_frag(Arow + k), false, load_frag_nt(Brow + k),
            (short)0, accI, false, false);

    v8f accS = {};
    const bool isE = (j0 < NE_);                 // block-uniform (NE multiple of 16)
    if (isE) {
        const unsigned short* Srow = aux   + (size_t)mn * NE_ + kofs;
        const unsigned short* Wrow = WtStp + (size_t)(j0 + mn) * NE_ + kofs;
#pragma unroll 8
        for (int k = eBeg; k < eEnd; k += 32)
            accS = __builtin_amdgcn_wmma_f32_16x16x32_bf16(
                false, load_frag(Srow + k), false, load_frag_nt(Wrow + k),
                (short)0, accS, false, false);
    }

    if (wave != 0) {
        *(v8f*)&red[wave - 1][0][lane * 8] = accE;
        *(v8f*)&red[wave - 1][1][lane * 8] = accI;
        *(v8f*)&red[wave - 1][2][lane * 8] = accS;
    }
    __syncthreads();
    if (wave != 0) return;

#pragma unroll
    for (int w = 0; w < 3; ++w) {
        accE += *(const v8f*)&red[w][0][lane * 8];
        accI += *(const v8f*)&red[w][1][lane * 8];
        accS += *(const v8f*)&red[w][2][lane * 8];
    }

    // Epilogue: C/D layout -> lane holds column j = j0 + (lane&15),
    // rows m = (lane>>4)*8 + i for i in 0..7.
    const int j = j0 + mn;
#pragma unroll
    for (int i = 0; i < 8; ++i) {
        const int b = kofs + i;
        const size_t sidx = (size_t)b * NN_ + j;
        const float hstp = isE ? accS[i] : 0.0f;
        const float hid  = accE[i] + accI[i] + hstp;  // hidden (+ h_stp on E cols)
        const float hid2 = accE[i] + hstp;            // hidden2 (+ h_stp on E cols)
        const float r0 = rec0[sidx] * C_EXP_SYN  + hid  * C_DT_SYN;
        const float r1 = rec1[sidx] * C_EXP_NMDA + hid2 * C_RN_DTN;
        const size_t fidx = ((size_t)b * TT_ + t) * NN_ + j;
        float nl = __builtin_nontemporal_load(ff + fidx) + r0 + r1 - C_THRESH;
        nl = nl > 0.f ? nl : 0.f;
        const float r = ratesf_in[sidx] * C_EXP_TAU + nl * C_DT_TAU;
        rec0[sidx] = r0;
        rec1[sidx] = r1;
        ratesf_out[sidx] = r;
        ratesb_out[sidx] = f32_to_bf16(r);
        __builtin_nontemporal_store(r, out + fidx);
    }
}

// ---------------- host launcher ----------------
extern "C" void kernel_launch(void* const* d_in, const int* in_sizes, int n_in,
                              void* d_out, int out_size, void* d_ws, size_t ws_size,
                              hipStream_t stream) {
    (void)in_sizes; (void)n_in; (void)out_size; (void)ws_size;
    const float* ff    = (const float*)d_in[0];  // (B,T,N)
    const float* rec   = (const float*)d_in[1];  // (2,B,N)
    const float* WabT  = (const float*)d_in[2];  // (N,N)
    const float* WstpT = (const float*)d_in[3];  // (NE,NE)
    float* out = (float*)d_out;                  // (B,T,N)

    char* ws = (char*)d_ws;
    auto carve = [&](size_t bytes) -> void* {
        void* p = (void*)ws; ws += (bytes + 255) & ~(size_t)255; return p;
    };
    unsigned short* WtMain = (unsigned short*)carve((size_t)NN_ * NN_ * 2);
    unsigned short* WtStp  = (unsigned short*)carve((size_t)NE_ * NE_ * 2);
    float* ratesf[2] = { (float*)carve((size_t)BB_ * NN_ * 4),
                         (float*)carve((size_t)BB_ * NN_ * 4) };
    unsigned short* ratesb[2] = { (unsigned short*)carve((size_t)BB_ * NN_ * 2),
                                  (unsigned short*)carve((size_t)BB_ * NN_ * 2) };
    float* rec0 = (float*)carve((size_t)BB_ * NN_ * 4);
    float* rec1 = (float*)carve((size_t)BB_ * NN_ * 4);
    float* u    = (float*)carve((size_t)BB_ * NE_ * 4);
    float* x    = (float*)carve((size_t)BB_ * NE_ * 4);
    unsigned short* aux = (unsigned short*)carve((size_t)BB_ * NE_ * 2);

    const dim3 tb(32, 8);
    transpose_to_bf16<<<dim3(NN_ / 32, NN_ / 32), tb, 0, stream>>>(WabT,  WtMain, NN_);
    transpose_to_bf16<<<dim3(NE_ / 32, NE_ / 32), tb, 0, stream>>>(WstpT, WtStp,  NE_);
    init_state<<<(BB_ * NN_ + 255) / 256, 256, 0, stream>>>(
        ff, rec, rec0, rec1, ratesf[0], ratesb[0], u, x);

    int cur = 0;
    for (int t = 0; t < TT_; ++t) {
        stp_update<<<(BB_ * NE_) / 256, 256, 0, stream>>>(ratesf[cur], u, x, aux);
        step_gemm<<<NN_ / 16, 128, 0, stream>>>(
            ratesb[cur], aux, WtMain, WtStp, rec0, rec1,
            ratesf[cur], ff, t, ratesf[cur ^ 1], ratesb[cur ^ 1], out);
        cur ^= 1;
    }
}
